// BiTRF_NLP_Crystal_13142599926298
// MI455X (gfx1250) — matrix operand order repaired and checked by reading, not compiled, not run on hardware
//
#include <hip/hip_runtime.h>
#include <hip/hip_bf16.h>

// ---------------------------------------------------------------------------
// B=2, L=1024, D=1024, H=16, DV=64, HID=4096, V=32000.
// attention scores = kproj[b,i,h] + posi[h,i,j]; kproj is constant along the
// softmax axis j and cancels -> attention weights are purely positional,
// W[h,i,j] = softmax_j(posi[h,i,j], diag masked), shared across batch.
// ---------------------------------------------------------------------------

typedef __attribute__((ext_vector_type(8)))  float   v8f;
typedef __attribute__((ext_vector_type(16))) __bf16  v16bf;
typedef __attribute__((ext_vector_type(8)))  __bf16  v8bf;

union BF16x16 {
    v16bf v;
    v8bf  h[2];
    unsigned int u[8];
};

// ---------------------------------------------------------------------------
// f32 -> bf16 conversion (grid-stride)
// ---------------------------------------------------------------------------
__global__ __launch_bounds__(256)
void cvt_f32_bf16_kernel(const float* __restrict__ s, __bf16* __restrict__ d, long n) {
    long i = (long)blockIdx.x * blockDim.x + threadIdx.x;
    long stride = (long)gridDim.x * blockDim.x;
    for (; i < n; i += stride) d[i] = (__bf16)s[i];
}

// ---------------------------------------------------------------------------
// block reduction helper (blockDim.x == 256)
// ---------------------------------------------------------------------------
__device__ __forceinline__ float block_reduce(float v, float* red, int is_max) {
    int t = threadIdx.x;
    red[t] = v;
    __syncthreads();
    for (int s = 128; s > 0; s >>= 1) {
        if (t < s) red[t] = is_max ? fmaxf(red[t], red[t + s]) : (red[t] + red[t + s]);
        __syncthreads();
    }
    float r = red[0];
    __syncthreads();
    return r;
}

__device__ __forceinline__ float softplusf(float x) {
    return (x > 20.f) ? x : log1pf(expf(x));
}

// ---------------------------------------------------------------------------
// Positional softmax: W[h,i,j] = softmax_j( p0*(e^{-sp1*|i-j|}+e^{-sp2*|i-j|}
//                               + p3*[i<j]) ; diag -> -inf )   (bf16 output)
// one block per (h,i) row; L == 1024, 256 threads x 4 elems
// ---------------------------------------------------------------------------
__global__ __launch_bounds__(256)
void posi_softmax_kernel(const float* __restrict__ p0, const float* __restrict__ p1,
                         const float* __restrict__ p2, const float* __restrict__ p3,
                         __bf16* __restrict__ W, int L) {
    __shared__ float red[256];
    int h = blockIdx.x / L;
    int i = blockIdx.x % L;
    float a0 = p0[h], a3 = p3[h];
    float sp1 = softplusf(p1[h]);
    float sp2 = softplusf(p2[h]);

    float vals[4];
    float vmax = -INFINITY;
    #pragma unroll
    for (int it = 0; it < 4; ++it) {
        int j = threadIdx.x + it * 256;
        float s = fabsf((float)(i - j));
        float v = a0 * (expf(-sp1 * s) + expf(-sp2 * s) + a3 * ((i < j) ? 1.f : 0.f));
        if (j == i) v = -INFINITY;
        vals[it] = v;
        vmax = fmaxf(vmax, v);
    }
    float m = block_reduce(vmax, red, 1);
    float sum = 0.f;
    #pragma unroll
    for (int it = 0; it < 4; ++it) {
        float e = expf(vals[it] - m);
        vals[it] = e;
        sum += e;
    }
    float tot = block_reduce(sum, red, 0);
    float inv = 1.f / tot;
    size_t base = ((size_t)h * L + i) * (size_t)L;
    #pragma unroll
    for (int it = 0; it < 4; ++it) {
        int j = threadIdx.x + it * 256;
        W[base + j] = (__bf16)(vals[it] * inv);
    }
}

// ---------------------------------------------------------------------------
// Generic bf16 WMMA GEMM:  C[m,n] = sum_k A[m,k] * B[n,k]  (+bias[n], opt ReLU)
// A: [M,K] bf16 row-major (lda), B: [N,K] bf16 row-major (ldb)
// C: f32 (optional), Cb: bf16 (optional), both [M,N] row-major (ldc)
// Per-z offsets:  off = (z / zmod) * s_div + (z % zmod) * s_mod
// Block: 128x128 tile; 8 waves (wave32); wave: 64x32 = 4x2 wmma tiles.
// Requires: M % 128 == 0, K % 64 == 0. N guarded by row-clamp (stores masked).
// Double-buffered K loop: fragments for k+32/k+64 stream while WMMAs for k run.
// ---------------------------------------------------------------------------
__global__ __launch_bounds__(256)
void gemm_bf16_nt_kernel(const __bf16* __restrict__ A, const __bf16* __restrict__ Bm,
                         float* __restrict__ C, __bf16* __restrict__ Cb,
                         const float* __restrict__ bias,
                         int N, int K, int lda, int ldb, int ldc,
                         int zmod,
                         long sAdiv, long sAmod, long sBdiv, long sBmod,
                         long sCdiv, long sCmod, int relu) {
    int z  = blockIdx.z;
    int zd = z / zmod;
    int zm = z % zmod;
    A  += zd * sAdiv + zm * sAmod;
    Bm += zd * sBdiv + zm * sBmod;
    long coff = zd * sCdiv + zm * sCmod;

    int lane = threadIdx.x & 31;
    int wave = threadIdx.x >> 5;
    int wm = wave >> 2;                 // 0..1
    int wn = wave & 3;                  // 0..3
    int m0 = blockIdx.y * 128 + wm * 64;
    int n0 = blockIdx.x * 128 + wn * 32;

    // A fragment (16-bit A 16x32): row = lane&15,
    //   elems 0..7 = K[k8..k8+7], elems 8..15 = K[k8+16..k8+23], k8=(lane>>4)*8
    int am = lane & 15;
    int ak = (lane >> 4) * 8;
    // B fragment (16-bit B 32x16): col = lane&15,
    //   elems 0..7 = K[kb..kb+7], elems 8..15 = K[kb+16..kb+23], kb=(lane>>4)*8
    int bn = lane & 15;
    int bk = (lane >> 4) * 8;

    // Per-lane base pointers. For B, clamp out-of-range columns to row N-1:
    // their garbage products land only in accumulator columns never stored.
    const __bf16* pA[4];
    #pragma unroll
    for (int mt = 0; mt < 4; ++mt)
        pA[mt] = A + (size_t)(m0 + mt * 16 + am) * lda + ak;
    const __bf16* pB[2];
    #pragma unroll
    for (int nt = 0; nt < 2; ++nt) {
        int nrow = n0 + nt * 16 + bn;
        if (nrow >= N) nrow = N - 1;
        pB[nt] = Bm + (size_t)nrow * ldb + bk;
    }

    v8f acc[4][2] = {};
    BF16x16 a0[4], b0[2], a1[4], b1[2];

    auto load = [&](BF16x16 (&af)[4], BF16x16 (&bf)[2], int k) {
        #pragma unroll
        for (int mt = 0; mt < 4; ++mt) {
            const __bf16* p = pA[mt] + k;
            af[mt].h[0] = *(const v8bf*)p;
            af[mt].h[1] = *(const v8bf*)(p + 16);
        }
        #pragma unroll
        for (int nt = 0; nt < 2; ++nt) {
            const __bf16* p = pB[nt] + k;
            bf[nt].h[0] = *(const v8bf*)p;
            bf[nt].h[1] = *(const v8bf*)(p + 16);
        }
    };
    auto mma = [&](BF16x16 (&af)[4], BF16x16 (&bf)[2]) {
        #pragma unroll
        for (int mt = 0; mt < 4; ++mt)
            #pragma unroll
            for (int nt = 0; nt < 2; ++nt)
                acc[mt][nt] = __builtin_amdgcn_wmma_f32_16x16x32_bf16(
                    false, af[mt].v, false, bf[nt].v,
                    (short)0, acc[mt][nt], false, false);
    };

    load(a0, b0, 0);
    int k = 0;
    for (; k + 64 < K; k += 64) {
        // stream ahead (speculative prefetch is safe past buffer end)
        __builtin_prefetch(pA[0] + k + 256, 0, 1);
        __builtin_prefetch(pA[2] + k + 256, 0, 1);
        __builtin_prefetch(pB[0] + k + 256, 0, 1);
        __builtin_prefetch(pB[1] + k + 256, 0, 1);
        load(a1, b1, k + 32);
        mma(a0, b0);
        load(a0, b0, k + 64);
        mma(a1, b1);
    }
    // tail: k == K-64
    load(a1, b1, k + 32);
    mma(a0, b0);
    mma(a1, b1);

    // Store: f32 C layout: lane 0-15 -> N=lane, M=r; lane 16-31 -> N=lane-16, M=r+8
    int col_lo = lane & 15;
    int row_hi = (lane >> 4) * 8;
    #pragma unroll
    for (int nt = 0; nt < 2; ++nt) {
        int col = n0 + nt * 16 + col_lo;
        if (col >= N) continue;
        float bvn = bias ? bias[col] : 0.f;
        #pragma unroll
        for (int mt = 0; mt < 4; ++mt) {
            #pragma unroll
            for (int r = 0; r < 8; ++r) {
                int row = m0 + mt * 16 + row_hi + r;
                float v = acc[mt][nt][r] + bvn;
                if (relu) v = fmaxf(v, 0.f);
                size_t idx = (size_t)coff + (size_t)row * ldc + col;
                if (C)  C[idx]  = v;
                if (Cb) Cb[idx] = (__bf16)v;
            }
        }
    }
}

// ---------------------------------------------------------------------------
// Repack V: vf32 [b,l,h,dv] f32 -> vt [b,h,dv,l] bf16
// ---------------------------------------------------------------------------
__global__ __launch_bounds__(256)
void repack_v_kernel(const float* __restrict__ vf, __bf16* __restrict__ vt,
                     int B, int L, int H, int DV) {
    long total = (long)B * L * H * DV;
    long i = (long)blockIdx.x * blockDim.x + threadIdx.x;
    long stride = (long)gridDim.x * blockDim.x;
    for (; i < total; i += stride) {
        long dv = i % DV;
        long t  = i / DV;
        long h  = t % H;
        long t2 = t / H;
        long l  = t2 % L;
        long b  = t2 / L;
        vt[(((b * H + h) * DV + dv) * L) + l] = (__bf16)vf[i];
    }
}

// ---------------------------------------------------------------------------
// LayerNorm over D=1024 per row; optional residual; writes f32 (opt) + bf16
// ---------------------------------------------------------------------------
__global__ __launch_bounds__(256)
void ln_kernel(const float* __restrict__ x, const float* __restrict__ res,
               const float* __restrict__ g, const float* __restrict__ b,
               float* __restrict__ outf, __bf16* __restrict__ outb, int D) {
    __shared__ float red[256];
    size_t base = (size_t)blockIdx.x * D;
    float vals[4];
    float s = 0.f;
    #pragma unroll
    for (int it = 0; it < 4; ++it) {
        int j = threadIdx.x + it * 256;
        float v = x[base + j] + (res ? res[base + j] : 0.f);
        vals[it] = v;
        s += v;
    }
    float mean = block_reduce(s, red, 0) / (float)D;
    float vs = 0.f;
    #pragma unroll
    for (int it = 0; it < 4; ++it) {
        float d = vals[it] - mean;
        vs += d * d;
    }
    float var = block_reduce(vs, red, 0) / (float)D;
    float rstd = rsqrtf(var + 1e-5f);
    #pragma unroll
    for (int it = 0; it < 4; ++it) {
        int j = threadIdx.x + it * 256;
        float o = (vals[it] - mean) * rstd * g[j] + b[j];
        if (outf) outf[base + j] = o;
        outb[base + j] = (__bf16)o;
    }
}

// ---------------------------------------------------------------------------
// log_softmax in place over V per row
// ---------------------------------------------------------------------------
__global__ __launch_bounds__(256)
void logsoftmax_kernel(float* __restrict__ p, int V) {
    __shared__ float red[256];
    float* row = p + (size_t)blockIdx.x * V;
    float m = -INFINITY;
    for (int j = threadIdx.x; j < V; j += 256) m = fmaxf(m, row[j]);
    m = block_reduce(m, red, 1);
    float s = 0.f;
    for (int j = threadIdx.x; j < V; j += 256) s += expf(row[j] - m);
    s = block_reduce(s, red, 0);
    float lse = m + logf(s);
    for (int j = threadIdx.x; j < V; j += 256) row[j] = row[j] - lse;
}

// ---------------------------------------------------------------------------
// Host launcher
// ---------------------------------------------------------------------------
extern "C" void kernel_launch(void* const* d_in, const int* in_sizes, int n_in,
                              void* d_out, int out_size, void* d_ws, size_t ws_size,
                              hipStream_t stream) {
    (void)in_sizes; (void)n_in; (void)out_size; (void)ws_size;
    constexpr int Bb = 2, L = 1024, D = 1024, H = 16, DV = 64, HID = 4096, V = 32000;
    constexpr int M = Bb * L;           // 2048 rows

    const float* x     = (const float*)d_in[0];
    // d_in[1]=wk, d_in[2]=bk: provably unused (cancel in softmax)
    const float* wv    = (const float*)d_in[3];
    const float* bv    = (const float*)d_in[4];
    const float* fc_w  = (const float*)d_in[5];
    const float* fc_b  = (const float*)d_in[6];
    const float* ln1_g = (const float*)d_in[7];
    const float* ln1_b = (const float*)d_in[8];
    const float* w1    = (const float*)d_in[9];
    const float* b1    = (const float*)d_in[10];
    const float* w2    = (const float*)d_in[11];
    const float* b2    = (const float*)d_in[12];
    const float* ln2_g = (const float*)d_in[13];
    const float* ln2_b = (const float*)d_in[14];
    const float* h2o_w = (const float*)d_in[15];
    const float* h2o_b = (const float*)d_in[16];
    const float* p0    = (const float*)d_in[17];
    const float* p1    = (const float*)d_in[18];
    const float* p2    = (const float*)d_in[19];
    const float* p3    = (const float*)d_in[20];
    float* logits = (float*)d_out;

    // bump allocator over workspace (256B aligned)
    char* ws = (char*)d_ws;
    size_t off = 0;
    auto alloc = [&](size_t bytes) -> void* {
        void* p = ws + off;
        off += (bytes + 255) & ~(size_t)255;
        return p;
    };
    __bf16* xb    = (__bf16*)alloc((size_t)M * D * 2);
    __bf16* wvb   = (__bf16*)alloc((size_t)H * DV * D * 2);
    __bf16* fcwb  = (__bf16*)alloc((size_t)D * H * DV * 2);
    __bf16* w1b   = (__bf16*)alloc((size_t)HID * D * 2);
    __bf16* w2b   = (__bf16*)alloc((size_t)D * HID * 2);
    __bf16* h2ob  = (__bf16*)alloc((size_t)V * D * 2);
    __bf16* Wb    = (__bf16*)alloc((size_t)H * L * L * 2);   // attention weights
    float*  vf32  = (float*) alloc((size_t)M * H * DV * 4);
    __bf16* vtb   = (__bf16*)alloc((size_t)M * H * DV * 2);  // [b,h,dv,l]
    float*  attnf = (float*) alloc((size_t)M * H * DV * 4);
    __bf16* attnb = (__bf16*)alloc((size_t)M * H * DV * 2);
    float*  fcout = (float*) alloc((size_t)M * D * 4);
    float*  ln1f  = (float*) alloc((size_t)M * D * 4);
    __bf16* ln1bb = (__bf16*)alloc((size_t)M * D * 2);
    __bf16* hmidb = (__bf16*)alloc((size_t)M * HID * 2);
    float*  w2out = vf32;                                    // vf32 dead by then
    float*  ln2f  = nullptr;                                 // not needed
    __bf16* ln2bb = (__bf16*)alloc((size_t)M * D * 2);

    auto cvt = [&](const float* s, __bf16* d, long n) {
        long blocks = (n + 1023) / 1024;
        if (blocks > 16384) blocks = 16384;
        cvt_f32_bf16_kernel<<<(int)blocks, 256, 0, stream>>>(s, d, n);
    };
    auto gemm = [&](const __bf16* A, const __bf16* Bm, float* C, __bf16* Cb,
                    const float* bias, int Mm, int N, int K,
                    int lda, int ldb, int ldc, int relu,
                    int gz = 1, int zmod = 1,
                    long sAd = 0, long sAm = 0, long sBd = 0, long sBm = 0,
                    long sCd = 0, long sCm = 0) {
        dim3 grid((N + 127) / 128, Mm / 128, gz);
        gemm_bf16_nt_kernel<<<grid, 256, 0, stream>>>(
            A, Bm, C, Cb, bias, N, K, lda, ldb, ldc,
            zmod, sAd, sAm, sBd, sBm, sCd, sCm, relu);
    };

    // 1) precision conversion of activations + weights
    cvt(x,     xb,   (long)M * D);
    cvt(wv,    wvb,  (long)H * DV * D);
    cvt(fc_w,  fcwb, (long)D * H * DV);
    cvt(w1,    w1b,  (long)HID * D);
    cvt(w2,    w2b,  (long)D * HID);
    cvt(h2o_w, h2ob, (long)V * D);

    // 2) positional attention weights W[h,i,j] (kproj cancels in softmax)
    posi_softmax_kernel<<<H * L, 256, 0, stream>>>(p0, p1, p2, p3, Wb, L);

    // 3) V projection: vf32[m, h*DV+dv] = x @ wv^T + bv
    gemm(xb, wvb, vf32, nullptr, bv, M, H * DV, D, D, D, H * DV, 0);
    repack_v_kernel<<<2048, 256, 0, stream>>>(vf32, vtb, Bb, L, H, DV);

    // 4) attention: per (b,h):  attnf[b, i, h*DV+dv] = W_h[i,:] @ V[b,:,h,dv]
    gemm(Wb, vtb, attnf, nullptr, nullptr, L, DV, L, L, L, H * DV, 0,
         /*gz*/ Bb * H, /*zmod*/ H,
         /*sAd*/ 0,                  /*sAm*/ (long)L * L,
         /*sBd*/ (long)H * DV * L,   /*sBm*/ (long)DV * L,
         /*sCd*/ (long)L * H * DV,   /*sCm*/ (long)DV);
    cvt(attnf, attnb, (long)M * H * DV);

    // 5) fc + LN1
    gemm(attnb, fcwb, fcout, nullptr, fc_b, M, D, H * DV, H * DV, H * DV, D, 0);
    ln_kernel<<<M, 256, 0, stream>>>(fcout, nullptr, ln1_g, ln1_b, ln1f, ln1bb, D);

    // 6) FFN: relu(ln1 @ w1^T + b1) @ w2^T + b2  (+ residual ln1) -> LN2
    gemm(ln1bb, w1b, nullptr, hmidb, b1, M, HID, D, D, D, HID, 1);
    gemm(hmidb, w2b, w2out, nullptr, b2, M, D, HID, HID, HID, D, 0);
    ln_kernel<<<M, 256, 0, stream>>>(w2out, ln1f, ln2_g, ln2_b, ln2f, ln2bb, D);

    // 7) vocab projection (dominant GEMM: 2048 x 32000 x 1024) + log_softmax
    gemm(ln2bb, h2ob, logits, nullptr, h2o_b, M, V, D, D, D, V, 0);
    logsoftmax_kernel<<<M, 256, 0, stream>>>(logits, V);
}
